// BertEmbedding_39316130628039
// MI455X (gfx1250) — matrix operand order repaired and checked by reading, compiled, not run on hardware
//
#include <hip/hip_runtime.h>
#include <math.h>

// ---------------- problem constants (match reference) ----------------
#define BB 16
#define SS 512
#define HH 768
#define LL 12
#define NHD 12
#define DH 64
#define FF_ 3072
#define WD_ 256
#define WW 255

// ---------------- WMMA types ----------------
typedef __attribute__((ext_vector_type(16))) __bf16 v16bf;
typedef __attribute__((ext_vector_type(8)))  float  v8f;
typedef __attribute__((ext_vector_type(4)))  unsigned int u32x4;

struct U256 { u32x4 lo; u32x4 hi; };          // 32 bytes -> v16bf
struct U16x16 { unsigned short s[16]; };       // 32 bytes -> v16bf

__device__ __forceinline__ unsigned short f2bf(float f) {
    unsigned int u = __builtin_bit_cast(unsigned int, f);
    u += 0x7FFFu + ((u >> 16) & 1u);           // round-to-nearest-even
    return (unsigned short)(u >> 16);
}

__device__ __forceinline__ v16bf ld_frag16B(const unsigned short* p0,
                                            const unsigned short* p1) {
    U256 t;
    t.lo = *(const u32x4*)p0;
    t.hi = *(const u32x4*)p1;
    return __builtin_bit_cast(v16bf, t);
}

// async 16B global -> LDS copy (CDNA5, tracked by ASYNCcnt)
__device__ __forceinline__ void async_cp16(const unsigned short* g, void* lds) {
    unsigned ldsoff = (unsigned)(unsigned long long)lds;   // low 32 bits = LDS offset
    asm volatile("global_load_async_to_lds_b128 %0, %1, off"
                 :: "v"(ldsoff), "v"(g) : "memory");
}
__device__ __forceinline__ void wait_async0() {
    asm volatile("s_wait_asynccnt 0x0" ::: "memory");
}

// =====================================================================
// WMMA GEMM: out = act( A[M,K](bf16) * Bt[N,K]^T(bf16) + bias + residual )
// 128x128 block tile, 8 waves (32x64 each), double-buffered LDS filled by
// global_load_async_to_lds_b128; copy of tile t+1 overlaps WMMA of tile t.
// act: 0 = none, 1 = exact GELU.  Cf (fp32) and Cb (bf16) are nullable.
// =====================================================================
__global__ __launch_bounds__(256) void gemm_bf16_wmma(
    const unsigned short* __restrict__ A,
    const unsigned short* __restrict__ Bt,
    const float* __restrict__ bias, const float* __restrict__ residual,
    float* __restrict__ Cf, unsigned short* __restrict__ Cb,
    int M, int N, int K, int act)
{
    __shared__ unsigned short As[2][128][40];   // [m][k], row pad 80B
    __shared__ unsigned short Bs[2][128][40];   // [n][k], row pad 80B

    const int tid  = threadIdx.x;
    const int wave = tid >> 5;
    const int lane = tid & 31;
    const int bm = blockIdx.y * 128;
    const int bn = blockIdx.x * 128;
    const int wm = (wave >> 1) * 32;
    const int wn = (wave & 1) * 64;
    const int lm = lane & 15;
    const int koffA = (lane < 16) ? 0 : 8;      // ISA 16-bit A layout K split
    const int koffB = (lane < 16) ? 0 : 8;      // B: same K split as A
    const int mAdd  = (lane < 16) ? 0 : 8;

    v8f acc[2][4] = {};

    // per thread: 2 rows-chunks of A + 2 of B per tile (4 async per wave)
    const int r0 = tid >> 2;                    // 0..63  -> two rows r0, r0+64
    const int c0 = (tid & 3) * 8;               // half offset within 32-half row

    auto issue_tile = [&](int kb, int buf) {
        #pragma unroll
        for (int h = 0; h < 2; ++h) {
            int r = r0 + h * 64;
            async_cp16(A  + (size_t)(bm + r) * K + kb + c0, &As[buf][r][c0]);
            async_cp16(Bt + (size_t)(bn + r) * K + kb + c0, &Bs[buf][r][c0]);
        }
    };

    const int nt = K >> 5;
    issue_tile(0, 0);

    for (int t = 0; t < nt; ++t) {
        const int buf = t & 1;
        wait_async0();                 // my copies for tile t landed in LDS
        __syncthreads();               // everyone's landed; prev buf free
        if (t + 1 < nt) issue_tile((t + 1) << 5, (t + 1) & 1);

        v16bf af[2];
        #pragma unroll
        for (int i = 0; i < 2; ++i) {
            int mrow = wm + i * 16 + lm;
            af[i] = ld_frag16B(&As[buf][mrow][koffA], &As[buf][mrow][16 + koffA]);
        }
        #pragma unroll
        for (int j = 0; j < 4; ++j) {
            int ncol = wn + j * 16 + lm;
            v16bf bf = ld_frag16B(&Bs[buf][ncol][koffB], &Bs[buf][ncol][koffB + 16]);
            #pragma unroll
            for (int i = 0; i < 2; ++i)
                acc[i][j] = __builtin_amdgcn_wmma_f32_16x16x32_bf16(
                    false, af[i], false, bf, (short)0, acc[i][j], false, false);
        }
    }

    // epilogue (C layout: lane n = lane&15, VGPR r -> m = r + (lane<16?0:8))
    #pragma unroll
    for (int j = 0; j < 4; ++j) {
        int n = bn + wn + j * 16 + lm;
        float bv = bias[n];
        #pragma unroll
        for (int i = 0; i < 2; ++i) {
            #pragma unroll
            for (int r = 0; r < 8; ++r) {
                int m = bm + wm + i * 16 + r + mAdd;
                float v = acc[i][j][r] + bv;
                if (residual) v += residual[(size_t)m * N + n];
                if (act == 1) v = 0.5f * v * (1.0f + erff(v * 0.70710678118f));
                if (Cf) Cf[(size_t)m * N + n] = v;
                if (Cb) Cb[(size_t)m * N + n] = f2bf(v);
            }
        }
    }
}

// =====================================================================
// Weight convert + transpose: in[K][N] fp32 -> out[N][K] bf16
// blockIdx.z selects the matrix (layer); 32x32 tiles, 32x8 threads.
// =====================================================================
__global__ __launch_bounds__(256) void transpose_bf16(
    const float* __restrict__ in, unsigned short* __restrict__ out, int K, int N)
{
    __shared__ unsigned short t[32][33];
    const int n0 = blockIdx.x * 32, k0 = blockIdx.y * 32;
    const float* inp = in + (size_t)blockIdx.z * K * N;
    unsigned short* outp = out + (size_t)blockIdx.z * K * N;
    for (int r = threadIdx.y; r < 32; r += 8)
        t[threadIdx.x][r] = f2bf(inp[(size_t)(k0 + r) * N + n0 + threadIdx.x]);
    __syncthreads();
    for (int r = threadIdx.y; r < 32; r += 8)
        outp[(size_t)(n0 + r) * K + k0 + threadIdx.x] = t[r][threadIdx.x];
}

// =====================================================================
// Fused attention per (b, head, 64-row q tile). bf16 in, bf16 ctx out.
// LDS: Qs[64][72], Ks[512][72] bf16; Ss[64][520] fp32; V^T reuses K region.
// 1/sqrt(DH) folded into the score epilogue.
// =====================================================================
#define ATTN_SMEM (64*72*2 + 512*72*2 + 64*520*4)   // 216064 bytes

__global__ __launch_bounds__(256) void attention_wmma(
    const unsigned short* __restrict__ q, const unsigned short* __restrict__ k,
    const unsigned short* __restrict__ v, const int* __restrict__ mask,
    unsigned short* __restrict__ ctx)
{
    extern __shared__ unsigned char smem_raw[];
    unsigned short (*Qs)[72]  = (unsigned short(*)[72])smem_raw;
    unsigned short (*Ks)[72]  = (unsigned short(*)[72])(smem_raw + 64*72*2);
    unsigned short (*Vt)[520] = (unsigned short(*)[520])(smem_raw + 64*72*2);
    float (*Ss)[520]          = (float(*)[520])(smem_raw + 64*72*2 + 512*72*2);

    const int tid  = threadIdx.x;
    const int wave = tid >> 5;
    const int lane = tid & 31;
    const int lm   = lane & 15;
    const int qt = blockIdx.x, hh = blockIdx.y, b = blockIdx.z;
    const int q0 = qt * 64;
    const int koffA = (lane < 16) ? 0 : 8;
    const int koffB = (lane < 16) ? 0 : 8;
    const int mAdd  = (lane < 16) ? 0 : 8;

    const unsigned short* qb = q + (size_t)(b * SS) * HH + hh * DH;
    const unsigned short* kp = k + (size_t)(b * SS) * HH + hh * DH;
    const unsigned short* vp = v + (size_t)(b * SS) * HH + hh * DH;
    const int* mk = mask + b * SS;

    // ---- stage Q tile and full K^T (16B vector copies) ----
    for (int i = tid; i < 64 * 8; i += 256) {        // 64 rows x 8 chunks
        int r = i >> 3, c = (i & 7) * 8;
        *(u32x4*)&Qs[r][c] = *(const u32x4*)(qb + (size_t)(q0 + r) * HH + c);
    }
    for (int i = tid; i < 512 * 8; i += 256) {       // 512 rows x 8 chunks
        int r = i >> 3, c = (i & 7) * 8;
        *(u32x4*)&Ks[r][c] = *(const u32x4*)(kp + (size_t)r * HH + c);
    }
    __syncthreads();

    // ---- scores: S[64][512] = Q * K^T, K-dim = 64 ----
    {
        const int mtile = wave & 3;
        const int nbase = (wave >> 2) * 16;
        for (int nc = 0; nc < 4; ++nc) {
            v8f acc[4] = {};
            #pragma unroll
            for (int ks = 0; ks < 2; ++ks) {
                int mrow = mtile * 16 + lm;
                v16bf aq = ld_frag16B(&Qs[mrow][ks*32 + koffA],
                                      &Qs[mrow][ks*32 + 16 + koffA]);
                #pragma unroll
                for (int j = 0; j < 4; ++j) {
                    int ncol = (nbase + nc * 4 + j) * 16 + lm;
                    v16bf bq = ld_frag16B(&Ks[ncol][ks*32 + koffB],
                                          &Ks[ncol][ks*32 + koffB + 16]);
                    acc[j] = __builtin_amdgcn_wmma_f32_16x16x32_bf16(
                        false, aq, false, bq, (short)0, acc[j], false, false);
                }
            }
            #pragma unroll
            for (int j = 0; j < 4; ++j) {
                int ncol = (nbase + nc * 4 + j) * 16 + lm;
                float bv = (mk[ncol] == 0) ? -10000.0f : 0.0f;
                #pragma unroll
                for (int r = 0; r < 8; ++r)
                    Ss[mtile * 16 + r + mAdd][ncol] = acc[j][r] * 0.125f + bv;
            }
        }
    }
    __syncthreads();

    // ---- softmax rows (64 threads) + stage V^T (all threads) ----
    if (tid < 64) {
        float mx = -3.4e38f;
        for (int j = 0; j < 512; ++j) mx = fmaxf(mx, Ss[tid][j]);
        float sum = 0.0f;
        for (int j = 0; j < 512; ++j) { float e = __expf(Ss[tid][j] - mx); Ss[tid][j] = e; sum += e; }
        float inv = 1.0f / sum;
        for (int j = 0; j < 512; ++j) Ss[tid][j] *= inv;
    }
    for (int i = tid; i < 512 * 64; i += 256) {
        int j = i >> 6, d = i & 63;
        Vt[d][j] = vp[(size_t)j * HH + d];           // transposed: [d][j]
    }
    __syncthreads();

    // ---- ctx: C[64][64] = P(64x512) * V(512x64), K-dim = 512 ----
    {
        const int mtile = wave >> 1;
        const int nt0   = (wave & 1) * 2;
        v8f acc2[2] = {};
        for (int ks = 0; ks < 16; ++ks) {
            const int kb0 = ks * 32;
            const int m = mtile * 16 + lm;
            U16x16 tu;
            #pragma unroll
            for (int vv = 0; vv < 4; ++vv) {          // e=0..7 -> K koffA..koffA+7
                tu.s[2*vv]     = f2bf(Ss[m][kb0 + koffA + 2*vv]);
                tu.s[2*vv + 1] = f2bf(Ss[m][kb0 + koffA + 2*vv + 1]);
            }
            #pragma unroll
            for (int vv = 0; vv < 4; ++vv) {          // e=8..15 -> K 16+koffA..
                tu.s[8 + 2*vv]     = f2bf(Ss[m][kb0 + 16 + koffA + 2*vv]);
                tu.s[8 + 2*vv + 1] = f2bf(Ss[m][kb0 + 16 + koffA + 2*vv + 1]);
            }
            v16bf ap = __builtin_bit_cast(v16bf, tu);
            #pragma unroll
            for (int j = 0; j < 2; ++j) {
                int ncol = (nt0 + j) * 16 + lm;
                v16bf bf = ld_frag16B(&Vt[ncol][kb0 + koffB],
                                      &Vt[ncol][kb0 + koffB + 16]);
                acc2[j] = __builtin_amdgcn_wmma_f32_16x16x32_bf16(
                    false, ap, false, bf, (short)0, acc2[j], false, false);
            }
        }
        #pragma unroll
        for (int j = 0; j < 2; ++j) {
            int d = (nt0 + j) * 16 + lm;
            #pragma unroll
            for (int r = 0; r < 8; ++r) {
                int m = mtile * 16 + r + mAdd;
                ctx[(size_t)(b * SS + q0 + m) * HH + hh * DH + d] = f2bf(acc2[j][r]);
            }
        }
    }
}

// =====================================================================
// Embeddings + LayerNorm (one block per token); fp32 + bf16 outputs
// =====================================================================
__global__ __launch_bounds__(256) void embed_ln(
    const int* __restrict__ ids, const float* __restrict__ wemb,
    const float* __restrict__ pemb, const float* __restrict__ temb,
    const float* __restrict__ s, const float* __restrict__ bp,
    float* __restrict__ out, unsigned short* __restrict__ outb)
{
    __shared__ float red[256];
    const int row = blockIdx.x;
    const int sp  = row & (SS - 1);
    const int tid = threadIdx.x;
    const int id  = ids[row];
    float a[3];
    #pragma unroll
    for (int i = 0; i < 3; ++i) {
        int d = tid + i * 256;
        a[i] = wemb[(size_t)id * HH + d] + pemb[(size_t)sp * HH + d] + temb[d];
    }
    red[tid] = a[0] + a[1] + a[2];
    __syncthreads();
    for (int o = 128; o > 0; o >>= 1) { if (tid < o) red[tid] += red[tid + o]; __syncthreads(); }
    float m = red[0] * (1.0f / HH);
    __syncthreads();
    float d0 = a[0] - m, d1 = a[1] - m, d2 = a[2] - m;
    red[tid] = d0 * d0 + d1 * d1 + d2 * d2;
    __syncthreads();
    for (int o = 128; o > 0; o >>= 1) { if (tid < o) red[tid] += red[tid + o]; __syncthreads(); }
    float r = rsqrtf(red[0] * (1.0f / HH) + 1e-12f);
    float v0 = d0 * r * s[tid]       + bp[tid];
    float v1 = d1 * r * s[tid + 256] + bp[tid + 256];
    float v2 = d2 * r * s[tid + 512] + bp[tid + 512];
    size_t base = (size_t)row * HH;
    out[base + tid] = v0; out[base + tid + 256] = v1; out[base + tid + 512] = v2;
    outb[base + tid] = f2bf(v0); outb[base + tid + 256] = f2bf(v1); outb[base + tid + 512] = f2bf(v2);
}

// =====================================================================
// LayerNorm over H=768 (one block per row); fp32 + bf16 outputs
// =====================================================================
__global__ __launch_bounds__(256) void layernorm_k(
    const float* __restrict__ x, const float* __restrict__ s,
    const float* __restrict__ bp, float* __restrict__ out,
    unsigned short* __restrict__ outb)
{
    __shared__ float red[256];
    const int row = blockIdx.x;
    const int tid = threadIdx.x;
    float a0 = x[(size_t)row * HH + tid];
    float a1 = x[(size_t)row * HH + tid + 256];
    float a2 = x[(size_t)row * HH + tid + 512];
    red[tid] = a0 + a1 + a2;
    __syncthreads();
    for (int o = 128; o > 0; o >>= 1) { if (tid < o) red[tid] += red[tid + o]; __syncthreads(); }
    float m = red[0] * (1.0f / HH);
    __syncthreads();
    float d0 = a0 - m, d1 = a1 - m, d2 = a2 - m;
    red[tid] = d0 * d0 + d1 * d1 + d2 * d2;
    __syncthreads();
    for (int o = 128; o > 0; o >>= 1) { if (tid < o) red[tid] += red[tid + o]; __syncthreads(); }
    float r = rsqrtf(red[0] * (1.0f / HH) + 1e-12f);
    float v0 = d0 * r * s[tid]       + bp[tid];
    float v1 = d1 * r * s[tid + 256] + bp[tid + 256];
    float v2 = d2 * r * s[tid + 512] + bp[tid + 512];
    size_t base = (size_t)row * HH;
    out[base + tid] = v0; out[base + tid + 256] = v1; out[base + tid + 512] = v2;
    outb[base + tid] = f2bf(v0); outb[base + tid + 256] = f2bf(v1); outb[base + tid + 512] = f2bf(v2);
}

// =====================================================================
// Ragged segment-mean merge: one block per (word, batch)
// =====================================================================
__global__ __launch_bounds__(256) void merge_k(
    const float* __restrict__ hid, const int* __restrict__ tlen,
    float* __restrict__ means, float* __restrict__ masks)
{
    __shared__ int sh[2];
    const int w = blockIdx.x, b = blockIdx.y;
    if (threadIdx.x == 0) {
        const int* tl = tlen + b * WW;
        int st = 0;
        for (int i = 0; i < w; ++i) st += tl[i];
        sh[0] = st; sh[1] = tl[w];
    }
    __syncthreads();
    const int start = sh[0], len = sh[1];
    const int d = threadIdx.x;               // WD == 256
    float ssum = 0.0f;
    for (int t = 0; t < len; ++t)
        ssum += hid[(size_t)(b * SS + 1 + start + t) * WD_ + d];
    means[((size_t)(b * WW) + w) * WD_ + d] = ssum / (float)(len > 0 ? len : 1);
    if (d == 0) masks[b * WW + w] = (len > 0) ? 1.0f : 0.0f;
}

// =====================================================================
// Host-side orchestration
// =====================================================================
extern "C" void kernel_launch(void* const* d_in, const int* in_sizes, int n_in,
                              void* d_out, int out_size, void* d_ws, size_t ws_size,
                              hipStream_t stream) {
    (void)in_sizes; (void)n_in; (void)out_size; (void)ws_size;
    const int*   input_ids = (const int*)d_in[0];
    const int*   attn_mask = (const int*)d_in[1];
    const int*   token_lens= (const int*)d_in[2];
    const float* word_emb  = (const float*)d_in[3];
    const float* pos_emb   = (const float*)d_in[4];
    const float* type_emb  = (const float*)d_in[5];
    const float* emb_ln_s  = (const float*)d_in[6];
    const float* emb_ln_b  = (const float*)d_in[7];
    const float* Wq = (const float*)d_in[8];  const float* bq = (const float*)d_in[9];
    const float* Wk = (const float*)d_in[10]; const float* bk = (const float*)d_in[11];
    const float* Wv = (const float*)d_in[12]; const float* bv = (const float*)d_in[13];
    const float* Wo = (const float*)d_in[14]; const float* bo = (const float*)d_in[15];
    const float* ln1s = (const float*)d_in[16]; const float* ln1b = (const float*)d_in[17];
    const float* W1 = (const float*)d_in[18]; const float* b1 = (const float*)d_in[19];
    const float* W2 = (const float*)d_in[20]; const float* b2 = (const float*)d_in[21];
    const float* ln2s = (const float*)d_in[22]; const float* ln2b = (const float*)d_in[23];
    const float* projW = (const float*)d_in[24]; const float* projb = (const float*)d_in[25];

    const int M = BB * SS;                    // 8192 rows
    char* ws = (char*)d_ws;
    size_t off = 0;
    const size_t szHf = (size_t)M * HH * sizeof(float);
    const size_t szHb = (size_t)M * HH * sizeof(unsigned short);
    float*          h    = (float*)(ws + off);          off += szHf;
    float*          tmp  = (float*)(ws + off);          off += szHf;
    float*          hid  = (float*)(ws + off);          off += (size_t)M * WD_ * sizeof(float);
    unsigned short* h_bf = (unsigned short*)(ws + off); off += szHb;
    unsigned short* q_bf = (unsigned short*)(ws + off); off += szHb;
    unsigned short* k_bf = (unsigned short*)(ws + off); off += szHb;
    unsigned short* v_bf = (unsigned short*)(ws + off); off += szHb;
    unsigned short* c_bf = (unsigned short*)(ws + off); off += szHb;
    unsigned short* f_bf = (unsigned short*)(ws + off); off += (size_t)M * FF_ * sizeof(unsigned short);
    const size_t szW  = (size_t)LL * HH * HH;
    const size_t szW1 = (size_t)LL * HH * FF_;
    unsigned short* Wqt = (unsigned short*)(ws + off); off += szW  * 2;
    unsigned short* Wkt = (unsigned short*)(ws + off); off += szW  * 2;
    unsigned short* Wvt = (unsigned short*)(ws + off); off += szW  * 2;
    unsigned short* Wot = (unsigned short*)(ws + off); off += szW  * 2;
    unsigned short* W1t = (unsigned short*)(ws + off); off += szW1 * 2;
    unsigned short* W2t = (unsigned short*)(ws + off); off += szW1 * 2;
    unsigned short* WPt = (unsigned short*)(ws + off); off += (size_t)HH * WD_ * 2;

    float* means = (float*)d_out;
    float* masks = means + (size_t)BB * WW * WD_;

    hipFuncSetAttribute((const void*)attention_wmma,
                        hipFuncAttributeMaxDynamicSharedMemorySize, ATTN_SMEM);

    const dim3 blk(256);
    const dim3 tb(32, 8);
    const dim3 gHH(HH / 128, M / 128);        // 6 x 64
    const dim3 gFF(FF_ / 128, M / 128);       // 24 x 64
    const dim3 gPJ(WD_ / 128, M / 128);       // 2 x 64
    const dim3 gAT(SS / 64, NHD, BB);         // 8 x 12 x 16

    // ---- weight convert + transpose (once per launch) ----
    transpose_bf16<<<dim3(HH/32, HH/32, LL), tb, 0, stream>>>(Wq, Wqt, HH, HH);
    transpose_bf16<<<dim3(HH/32, HH/32, LL), tb, 0, stream>>>(Wk, Wkt, HH, HH);
    transpose_bf16<<<dim3(HH/32, HH/32, LL), tb, 0, stream>>>(Wv, Wvt, HH, HH);
    transpose_bf16<<<dim3(HH/32, HH/32, LL), tb, 0, stream>>>(Wo, Wot, HH, HH);
    transpose_bf16<<<dim3(FF_/32, HH/32, LL), tb, 0, stream>>>(W1, W1t, HH, FF_);
    transpose_bf16<<<dim3(HH/32, FF_/32, LL), tb, 0, stream>>>(W2, W2t, FF_, HH);
    transpose_bf16<<<dim3(WD_/32, HH/32, 1),  tb, 0, stream>>>(projW, WPt, HH, WD_);

    embed_ln<<<M, blk, 0, stream>>>(input_ids, word_emb, pos_emb, type_emb,
                                    emb_ln_s, emb_ln_b, h, h_bf);

    for (int l = 0; l < LL; ++l) {
        const unsigned short* Wqt_l = Wqt + (size_t)l * HH * HH;
        const unsigned short* Wkt_l = Wkt + (size_t)l * HH * HH;
        const unsigned short* Wvt_l = Wvt + (size_t)l * HH * HH;
        const unsigned short* Wot_l = Wot + (size_t)l * HH * HH;
        const unsigned short* W1t_l = W1t + (size_t)l * HH * FF_;
        const unsigned short* W2t_l = W2t + (size_t)l * HH * FF_;
        const float* bq_l = bq + (size_t)l * HH;
        const float* bk_l = bk + (size_t)l * HH;
        const float* bv_l = bv + (size_t)l * HH;
        const float* bo_l = bo + (size_t)l * HH;
        const float* b1_l = b1 + (size_t)l * FF_;
        const float* b2_l = b2 + (size_t)l * HH;

        gemm_bf16_wmma<<<gHH, blk, 0, stream>>>(h_bf, Wqt_l, bq_l, nullptr, nullptr, q_bf, M, HH, HH, 0);
        gemm_bf16_wmma<<<gHH, blk, 0, stream>>>(h_bf, Wkt_l, bk_l, nullptr, nullptr, k_bf, M, HH, HH, 0);
        gemm_bf16_wmma<<<gHH, blk, 0, stream>>>(h_bf, Wvt_l, bv_l, nullptr, nullptr, v_bf, M, HH, HH, 0);

        attention_wmma<<<gAT, blk, ATTN_SMEM, stream>>>(q_bf, k_bf, v_bf, attn_mask, c_bf);

        gemm_bf16_wmma<<<gHH, blk, 0, stream>>>(c_bf, Wot_l, bo_l, h, tmp, nullptr, M, HH, HH, 0);
        layernorm_k<<<M, blk, 0, stream>>>(tmp, ln1s + (size_t)l * HH, ln1b + (size_t)l * HH, h, h_bf);

        gemm_bf16_wmma<<<gFF, blk, 0, stream>>>(h_bf, W1t_l, b1_l, nullptr, nullptr, f_bf, M, FF_, HH, 1);
        gemm_bf16_wmma<<<gHH, blk, 0, stream>>>(f_bf, W2t_l, b2_l, h, tmp, nullptr, M, HH, FF_, 0);
        layernorm_k<<<M, blk, 0, stream>>>(tmp, ln2s + (size_t)l * HH, ln2b + (size_t)l * HH, h, h_bf);
    }

    gemm_bf16_wmma<<<gPJ, blk, 0, stream>>>(h_bf, WPt, projb, nullptr, hid, nullptr, M, WD_, HH, 0);

    merge_k<<<dim3(WW, BB), blk, 0, stream>>>(hid, token_lens, means, masks);
}
